// CentroidDiscoverBlock_32976758898704
// MI455X (gfx1250) — matrix operand order repaired and checked by reading, not compiled, not run on hardware
//
#include <hip/hip_runtime.h>
#include <math.h>

// ---- problem constants (from reference) ----
#define Bn  4
#define Tn  8
#define Nn  4096
#define Ln  64
#define Cn  256
#define Dn  256
#define ROWS (Bn*Tn*Nn)            // 131072 node rows
#define RPB  (Tn*Nn)               // 32768 rows per batch
#define XPITCH 264                 // padded f16 LDS row pitch (8-half pad -> conflict-free b128)

typedef __attribute__((ext_vector_type(16))) _Float16 v16h;
typedef __attribute__((ext_vector_type(8)))  _Float16 v8h;
typedef __attribute__((ext_vector_type(8)))  float    v8f;

// ---------------- init: zero segment-sum buffers ----------------
__global__ void k_init(float* sums, float* counts) {
  int t = blockIdx.x * 256 + threadIdx.x;
  if (t < Bn*Ln*Cn) sums[t] = 0.f;
  if (t < Bn*Ln)    counts[t] = 0.f;
}

// ---------------- Qc = centroids @ qc_w^T + qc_b  [B,L,C] ----------------
__global__ void k_qcent(const float* __restrict__ ct, const float* __restrict__ qc_w,
                        const float* __restrict__ qc_b, float* __restrict__ Qc) {
  int t = blockIdx.x * 256 + threadIdx.x;        // B*L*C = 65536
  int c = t & 255, bl = t >> 8;
  const float* x = ct + bl * Dn;
  const float* w = qc_w + c * Dn;
  float a = qc_b[c];
  for (int d = 0; d < Dn; ++d) a += x[d] * w[d];
  Qc[t] = a;
}

// ---------------- bias2[b,l] = nk_b . Qc[b,l] ----------------
__global__ void k_bias2(const float* __restrict__ Qc, const float* __restrict__ nk_b,
                        float* __restrict__ bias2) {
  int t = threadIdx.x;                           // 256 = B*L
  const float* q = Qc + t * Cn;
  float a = 0.f;
  for (int c = 0; c < Cn; ++c) a += nk_b[c] * q[c];
  bias2[t] = a;
}

// ---- pack A2[b,l,i] = sum_c Qc[b,l,c]*nk_w[c,i] into WMMA B-fragment order (f16) ----
// B-frag layout: lane = (n&15) + 16*hi, hi = (k_in>>3)&1; element e = (k_in&7) + 8*(k_in>>4)
// (K {0..7,16..23} for hi=0, {8..15,24..31} for hi=1). packed[((b*4+nt)*8+ks)*32+lane][16]
__global__ void k_prepA2(const float* __restrict__ Qc, const float* __restrict__ nk_w,
                         _Float16* __restrict__ pA2) {
  int t = blockIdx.x * 256 + threadIdx.x;        // B*L*C = 65536
  int i = t & 255, bl = t >> 8, b = bl >> 6, l = bl & 63;
  const float* q = Qc + bl * Cn;
  float a = 0.f;
  for (int c = 0; c < Cn; ++c) a += q[c] * nk_w[c * Cn + i];
  int ks = i >> 5, kin = i & 31, hi = (kin >> 3) & 1, e = (kin & 7) | ((kin >> 4) << 3);
  int nt = l >> 4, lane = (l & 15) + (hi << 4);
  pA2[((((b * 4 + nt) * 8 + ks) * 32 + lane) << 4) + e] = (_Float16)a;
}

// ---- pack nv_w [out=256, in=256] into WMMA B-fragment order (f16) ----
__global__ void k_packWv(const float* __restrict__ wv, _Float16* __restrict__ pWv) {
  int t = blockIdx.x * 256 + threadIdx.x;        // 65536
  int i = t & 255, n = t >> 8;
  int ks = i >> 5, kin = i & 31, hi = (kin >> 3) & 1, e = (kin & 7) | ((kin >> 4) << 3);
  int nt = n >> 4, lane = (n & 15) + (hi << 4);
  pWv[(((nt * 8 + ks) * 32 + lane) << 4) + e] = (_Float16)(wv[n * Cn + i]);
}

// ---------------- fused main kernel: scores GEMM + argmax + V GEMM + binning ----------------
__launch_bounds__(256)
__global__ void k_main(const float* __restrict__ X, const _Float16* __restrict__ pA2,
                       const float* __restrict__ bias2, const _Float16* __restrict__ pWv,
                       float* __restrict__ sums, float* __restrict__ counts) {
  extern __shared__ char smem[];
  _Float16* xs = (_Float16*)smem;                                  // 128*264 f16 = 67584 B
  float* sc    = (float*)(smem + 128 * XPITCH * 2);                // 128*64 f32 = 32768 B
  float* vb    = (float*)(smem + 128 * XPITCH * 2 + 128 * 64 * 4); // 64*256 f32 = 65536 B
  int* assignb = (int*)(smem + 128 * XPITCH * 2 + 128 * 64 * 4 + Ln * Cn * 4); // 128 ints
  int* hist    = assignb + 128;                                    // 64 ints

  const int tid  = threadIdx.x;
  const int wave = tid >> 5, lane = tid & 31;
  const int lrow = lane & 15, hi = lane >> 4;
  const int blk  = blockIdx.x;
  const int b    = blk >> 6;          // 64 blocks (512 rows each) per batch
  const int brow0 = blk * 512;

  for (int i = tid; i < Ln * Cn; i += 256) vb[i] = 0.f;
  if (tid < Ln) hist[tid] = 0;
  __syncthreads();

  const _Float16* pA2b = pA2 + b * (4 * 8 * 32 * 16);
  const float* b2 = bias2 + b * Ln;
  const v8f vzero = {0.f, 0.f, 0.f, 0.f, 0.f, 0.f, 0.f, 0.f};

  for (int sub = 0; sub < 4; ++sub) {
    const int row0 = brow0 + sub * 128;
    // ---- stage 128x256 f32 -> f16 LDS tile ----
    const float* Xb = X + (size_t)row0 * Cn;
    for (int idx = tid; idx < 128 * 64; idx += 256) {
      int r = idx >> 6, c4 = (idx & 63) << 2;
      const float4 f = *(const float4*)(Xb + r * Cn + c4);
      int o = r * XPITCH + c4;
      xs[o + 0] = (_Float16)f.x; xs[o + 1] = (_Float16)f.y;
      xs[o + 2] = (_Float16)f.z; xs[o + 3] = (_Float16)f.w;
    }
    __syncthreads();

    const int wrow = wave * 16;
    const _Float16* arowp = xs + (wrow + lrow) * XPITCH + hi * 8;

    // ---- scores = X @ A2^T  (+bias2), 64 outputs ----
    v8f acc[4];
#pragma unroll
    for (int n = 0; n < 4; ++n) acc[n] = vzero;
#pragma unroll
    for (int ks = 0; ks < 8; ++ks) {
      union { v16h v; v8h h[2]; } A;
      const _Float16* ap = arowp + ks * 32;
      A.h[0] = *(const v8h*)(ap);          // K {0..7} or {8..15}
      A.h[1] = *(const v8h*)(ap + 16);     // K {16..23} or {24..31}
#pragma unroll
      for (int nt = 0; nt < 4; ++nt) {
        v16h Bv = *(const v16h*)(pA2b + (((nt * 8 + ks) * 32 + lane) << 4));
        acc[nt] = __builtin_amdgcn_wmma_f32_16x16x32_f16(
            false, A.v, false, Bv, (short)0, acc[nt], false, false);
      }
    }
#pragma unroll
    for (int nt = 0; nt < 4; ++nt) {
      int col = nt * 16 + lrow;
      float bb = b2[col];
#pragma unroll
      for (int r = 0; r < 8; ++r) {
        int row = wrow + r + (hi << 3);
        sc[row * 64 + col] = acc[nt][r] + bb;
      }
    }
    __syncthreads();

    // ---- per-row argmax over L=64 ----
    if (tid < 128) {
      const float* s = sc + tid * 64;
      float best = s[0]; int bi = 0;
      for (int l = 1; l < Ln; ++l) { float v = s[l]; if (v > best) { best = v; bi = l; } }
      assignb[tid] = bi;
      atomicAdd(&hist[bi], 1);
    }
    __syncthreads();

    // ---- V = X @ nv_w^T, binned into LDS by assignment ----
#pragma unroll 1
    for (int g = 0; g < 4; ++g) {
      v8f vac[4];
#pragma unroll
      for (int n = 0; n < 4; ++n) vac[n] = vzero;
#pragma unroll
      for (int ks = 0; ks < 8; ++ks) {
        union { v16h v; v8h h[2]; } A;
        const _Float16* ap = arowp + ks * 32;
        A.h[0] = *(const v8h*)(ap);
        A.h[1] = *(const v8h*)(ap + 16);
#pragma unroll
        for (int nt = 0; nt < 4; ++nt) {
          int gnt = g * 4 + nt;
          v16h Bv = *(const v16h*)(pWv + (((gnt * 8 + ks) * 32 + lane) << 4));
          vac[nt] = __builtin_amdgcn_wmma_f32_16x16x32_f16(
              false, A.v, false, Bv, (short)0, vac[nt], false, false);
        }
      }
#pragma unroll
      for (int nt = 0; nt < 4; ++nt) {
#pragma unroll
        for (int r = 0; r < 8; ++r) {
          int row = wrow + r + (hi << 3);
          int l = assignb[row];
          int col = g * 64 + nt * 16 + lrow;
          atomicAdd(&vb[l * Cn + col], vac[nt][r]);
        }
      }
    }
    __syncthreads();
  }

  // ---- flush bins to global (L2-resident 256KB buffer) ----
  float* sb = sums + b * Ln * Cn;
  for (int i = tid; i < Ln * Cn; i += 256) atomicAdd(&sb[i], vb[i]);
  if (tid < Ln) atomicAdd(&counts[b * Ln + tid], (float)hist[tid]);
}

// ---------------- cent = centroids + (cluster @ al_w^T + al_b) ----------------
// cluster[c] = (sums[c] + cnt*nv_b[c]) / (cnt^2 + 1)   (V bias folded: count * nv_b)
__global__ void k_cent(const float* __restrict__ ct, const float* __restrict__ sums,
                       const float* __restrict__ counts, const float* __restrict__ nv_b,
                       const float* __restrict__ al_w, const float* __restrict__ al_b,
                       float* __restrict__ cent) {
  int t = blockIdx.x * 256 + threadIdx.x;        // B*L*D = 65536
  int d = t & 255, bl = t >> 8;
  float cnt = counts[bl];
  float inv = 1.f / (cnt * cnt + 1.f);
  const float* s = sums + bl * Cn;
  const float* w = al_w + d * Cn;
  float a = 0.f;
  for (int c = 0; c < Cn; ++c) a += (s[c] + cnt * nv_b[c]) * w[c];
  cent[t] = ct[t] + al_b[d] + inv * a;
}

// ---------------- q/k/v projections ----------------
__global__ void k_qkv(const float* __restrict__ cent,
                      const float* __restrict__ wq, const float* __restrict__ bq,
                      const float* __restrict__ wk, const float* __restrict__ bk,
                      const float* __restrict__ wv, const float* __restrict__ bv,
                      float* __restrict__ q, float* __restrict__ k, float* __restrict__ v) {
  int t = blockIdx.x * 256 + threadIdx.x;
  int d = t & 255, bl = t >> 8;
  const float *w, *bb; float* o;
  if (blockIdx.y == 0)      { w = wq; bb = bq; o = q; }
  else if (blockIdx.y == 1) { w = wk; bb = bk; o = k; }
  else                      { w = wv; bb = bv; o = v; }
  const float* x = cent + bl * Dn;
  const float* wr = w + d * Dn;
  float a = bb[d];
  for (int c = 0; c < Dn; ++c) a += x[c] * wr[c];
  o[t] = a;
}

// ---------------- per-(b,h) softmax attention, L=64, hd=64 ----------------
__global__ void k_attn(const float* __restrict__ q, const float* __restrict__ k,
                       const float* __restrict__ v, float* __restrict__ attn_out) {
  __shared__ float ks_[64][65];
  __shared__ float vs_[64][65];
  __shared__ float sc_[64][65];
  int bh = blockIdx.x, b = bh >> 2, h = bh & 3;
  int t = threadIdx.x;                           // 64 threads, one query row each
  for (int i = t; i < 64 * 64; i += 64) {
    int row = i >> 6, e = i & 63;
    ks_[row][e] = k[(b * Ln + row) * Dn + h * 64 + e];
    vs_[row][e] = v[(b * Ln + row) * Dn + h * 64 + e];
  }
  __syncthreads();
  float qreg[64];
#pragma unroll
  for (int e = 0; e < 64; ++e) qreg[e] = q[(b * Ln + t) * Dn + h * 64 + e];
  for (int kl = 0; kl < 64; ++kl) {
    float s = 0.f;
#pragma unroll
    for (int e = 0; e < 64; ++e) s += qreg[e] * ks_[kl][e];
    sc_[t][kl] = s * 0.125f;                     // 1/sqrt(64)
  }
  float mx = sc_[t][0];
  for (int kl = 1; kl < 64; ++kl) mx = fmaxf(mx, sc_[t][kl]);
  float den = 0.f;
  for (int kl = 0; kl < 64; ++kl) { float p = expf(sc_[t][kl] - mx); sc_[t][kl] = p; den += p; }
  float dinv = 1.f / den;
  float o[64];
#pragma unroll
  for (int e = 0; e < 64; ++e) o[e] = 0.f;
  for (int kl = 0; kl < 64; ++kl) {
    float p = sc_[t][kl] * dinv;
#pragma unroll
    for (int e = 0; e < 64; ++e) o[e] += p * vs_[kl][e];
  }
#pragma unroll
  for (int e = 0; e < 64; ++e) attn_out[(b * Ln + t) * Dn + h * 64 + e] = o[e];
}

// ---------------- z2 = cent + attn_out @ mo_w^T + mo_b ----------------
__global__ void k_z2(const float* __restrict__ cent, const float* __restrict__ attn_out,
                     const float* __restrict__ mo_w, const float* __restrict__ mo_b,
                     float* __restrict__ z2) {
  int t = blockIdx.x * 256 + threadIdx.x;
  int d = t & 255, bl = t >> 8;
  const float* x = attn_out + bl * Dn;
  const float* w = mo_w + d * Dn;
  float a = mo_b[d];
  for (int c = 0; c < Dn; ++c) a += x[c] * w[c];
  z2[t] = cent[t] + a;
}

// ---------------- batchnorm stats over (B,L)=256 samples per channel ----------------
__global__ void k_bnstats(const float* __restrict__ z2, float* __restrict__ mean,
                          float* __restrict__ istd) {
  int d = threadIdx.x;                           // 256 channels, one block
  float s = 0.f, s2 = 0.f;
  for (int r = 0; r < Bn * Ln; ++r) { float x = z2[r * Dn + d]; s += x; s2 += x * x; }
  float m = s * (1.f / (Bn * Ln));
  float var = s2 * (1.f / (Bn * Ln)) - m * m;
  mean[d] = m;
  istd[d] = rsqrtf(var + 1e-5f);
}

// ---------------- zn = alpha*(gamma*(z2-mean)*istd + beta) + bias ----------------
__global__ void k_zn(const float* __restrict__ z2, const float* __restrict__ mean,
                     const float* __restrict__ istd, const float* __restrict__ gamma,
                     const float* __restrict__ beta, const float* __restrict__ alpha,
                     const float* __restrict__ bias, float* __restrict__ zn) {
  int t = blockIdx.x * 256 + threadIdx.x;
  int d = t & 255;
  float n = (z2[t] - mean[d]) * istd[d];
  zn[t] = alpha[d] * (gamma[d] * n + beta[d]) + bias[d];
}

// ---------------- h1 = relu(zn @ ff1_w^T + ff1_b) ----------------
__global__ void k_ff1(const float* __restrict__ zn, const float* __restrict__ w,
                      const float* __restrict__ bb, float* __restrict__ h1) {
  int t = blockIdx.x * 256 + threadIdx.x;
  int d = t & 255, bl = t >> 8;
  const float* x = zn + bl * Dn;
  const float* wr = w + d * Dn;
  float a = bb[d];
  for (int c = 0; c < Dn; ++c) a += x[c] * wr[c];
  h1[t] = fmaxf(a, 0.f);
}

// ---------------- out = h1 @ ff2_w^T + ff2_b ----------------
__global__ void k_out(const float* __restrict__ h1, const float* __restrict__ w,
                      const float* __restrict__ bb, float* __restrict__ out) {
  int t = blockIdx.x * 256 + threadIdx.x;
  int d = t & 255, bl = t >> 8;
  const float* x = h1 + bl * Dn;
  const float* wr = w + d * Dn;
  float a = bb[d];
  for (int c = 0; c < Dn; ++c) a += x[c] * wr[c];
  out[t] = a;
}

extern "C" void kernel_launch(void* const* d_in, const int* in_sizes, int n_in,
                              void* d_out, int out_size, void* d_ws, size_t ws_size,
                              hipStream_t stream) {
  const float* STF  = (const float*)d_in[0];
  const float* ct   = (const float*)d_in[1];
  const float* qc_w = (const float*)d_in[2];  const float* qc_b = (const float*)d_in[3];
  const float* nk_w = (const float*)d_in[4];  const float* nk_b = (const float*)d_in[5];
  const float* nv_w = (const float*)d_in[6];  const float* nv_b = (const float*)d_in[7];
  const float* al_w = (const float*)d_in[8];  const float* al_b = (const float*)d_in[9];
  const float* mq_w = (const float*)d_in[10]; const float* mq_b = (const float*)d_in[11];
  const float* mk_w = (const float*)d_in[12]; const float* mk_b = (const float*)d_in[13];
  const float* mv_w = (const float*)d_in[14]; const float* mv_b = (const float*)d_in[15];
  const float* mo_w = (const float*)d_in[16]; const float* mo_b = (const float*)d_in[17];
  const float* bn_g = (const float*)d_in[18]; const float* bn_b = (const float*)d_in[19];
  const float* alp  = (const float*)d_in[20]; const float* bia  = (const float*)d_in[21];
  const float* f1w  = (const float*)d_in[22]; const float* f1b  = (const float*)d_in[23];
  const float* f2w  = (const float*)d_in[24]; const float* f2b  = (const float*)d_in[25];
  float* out = (float*)d_out;

  // workspace bump allocator (256B aligned)
  char* wsb = (char*)d_ws;
  size_t off = 0;
  auto alloc = [&](size_t bytes) -> char* {
    char* p = wsb + off;
    off = (off + bytes + 255) & ~(size_t)255;
    return p;
  };
  float*    Qc     = (float*)   alloc(Bn * Ln * Cn * 4);
  float*    bias2  = (float*)   alloc(Bn * Ln * 4);
  _Float16* pA2    = (_Float16*)alloc(Bn * 4 * 8 * 32 * 16 * 2);
  _Float16* pWv    = (_Float16*)alloc(16 * 8 * 32 * 16 * 2);
  float*    sums   = (float*)   alloc(Bn * Ln * Cn * 4);
  float*    counts = (float*)   alloc(Bn * Ln * 4);
  float*    cent   = (float*)   alloc(Bn * Ln * Dn * 4);
  float*    qb     = (float*)   alloc(Bn * Ln * Dn * 4);
  float*    kb     = (float*)   alloc(Bn * Ln * Dn * 4);
  float*    vb     = (float*)   alloc(Bn * Ln * Dn * 4);
  float*    attn   = (float*)   alloc(Bn * Ln * Dn * 4);
  float*    z2     = (float*)   alloc(Bn * Ln * Dn * 4);
  float*    meanb  = (float*)   alloc(Dn * 4);
  float*    istdb  = (float*)   alloc(Dn * 4);
  float*    znb    = (float*)   alloc(Bn * Ln * Dn * 4);
  float*    h1     = (float*)   alloc(Bn * Ln * Dn * 4);
  (void)ws_size; (void)in_sizes; (void)n_in; (void)out_size;

  k_init  <<<256, 256, 0, stream>>>(sums, counts);
  k_qcent <<<256, 256, 0, stream>>>(ct, qc_w, qc_b, Qc);
  k_bias2 <<<1, 256, 0, stream>>>(Qc, nk_b, bias2);
  k_prepA2<<<256, 256, 0, stream>>>(Qc, nk_w, pA2);
  k_packWv<<<256, 256, 0, stream>>>(nv_w, pWv);

  // LDS: X tile (67584) + scores (32768) + bins (65536) + assign/hist (768)
  size_t smem = 128 * XPITCH * 2 + 128 * 64 * 4 + Ln * Cn * 4 + 128 * 4 + Ln * 4;
  k_main<<<ROWS / 512, 256, smem, stream>>>(STF, pA2, bias2, pWv, sums, counts);

  k_cent<<<256, 256, 0, stream>>>(ct, sums, counts, nv_b, al_w, al_b, cent);
  k_qkv <<<dim3(256, 3), 256, 0, stream>>>(cent, mq_w, mq_b, mk_w, mk_b, mv_w, mv_b, qb, kb, vb);
  k_attn<<<Bn * 4, 64, 0, stream>>>(qb, kb, vb, attn);
  k_z2  <<<256, 256, 0, stream>>>(cent, attn, mo_w, mo_b, z2);
  k_bnstats<<<1, 256, 0, stream>>>(z2, meanb, istdb);
  k_zn  <<<256, 256, 0, stream>>>(z2, meanb, istdb, bn_g, bn_b, alp, bia, znb);
  k_ff1 <<<256, 256, 0, stream>>>(znb, f1w, f1b, h1);
  k_out <<<256, 256, 0, stream>>>(h1, f2w, f2b, out);
}
